// NetWithEncoderDecoderGRU_properRL_37443524886773
// MI455X (gfx1250) — compile-verified, matrix operand order checked by reading, not run on hardware
//
#include <hip/hip_runtime.h>

// B=16384, E=128, H=256, OUT=2, S=32, CAR=8
#define B_TOT    16384
#define EDIM     128
#define HDIM     256
#define SDEC     32
#define BT       64          // batch rows per workgroup
#define HP       264         // padded LDS row stride (bf16 elems): 132 dwords -> 4-bank rotation
#define NTHREADS 256

typedef __bf16 bf16;
typedef __attribute__((ext_vector_type(16))) __bf16 v16bf;
typedef __attribute__((ext_vector_type(8)))  __bf16 v8bf;
typedef __attribute__((ext_vector_type(8)))  float  v8f;

// Load a 16x32 bf16 A/B WMMA fragment slice: this lane's 16 contiguous bf16 (32B) as 2x16B.
__device__ __forceinline__ v16bf ldfrag(const bf16* p) {
    union { v16bf v; v8bf h[2]; } f;
    f.h[0] = *(const v8bf*)(p);
    f.h[1] = *(const v8bf*)(p + 8);
    return f.v;
}

__device__ __forceinline__ v8f wmma_bf(v16bf a, v16bf b, v8f c) {
    // (neg_a, A, neg_b, B, c_mod, C, reuse_a, reuse_b)
    return __builtin_amdgcn_wmma_f32_16x16x32_bf16(false, a, false, b, (short)0, c, false, false);
}

__device__ __forceinline__ float sigm_(float x)  { return 1.0f / (1.0f + __expf(-x)); }
__device__ __forceinline__ float tanhf_(float x) { float e = __expf(2.0f * x); return (e - 1.0f) / (e + 1.0f); }

__global__ void cvt_bf16_kernel(const float* __restrict__ s, bf16* __restrict__ d, int n) {
    int i = blockIdx.x * blockDim.x + threadIdx.x;
    if (i < n) d[i] = (bf16)s[i];
}

__global__ __launch_bounds__(NTHREADS)
void gru_net_kernel(const float* __restrict__ x,
                    const float* __restrict__ emb_mt_W, const float* __restrict__ emb_mt_b,
                    const float* __restrict__ emb_v_W,  const float* __restrict__ emb_v_b,
                    const bf16*  __restrict__ encWih,   const bf16*  __restrict__ encWhh,
                    const float* __restrict__ enc_bih,  const float* __restrict__ enc_bhh,
                    const bf16*  __restrict__ decWih,   const bf16*  __restrict__ decWhh,
                    const float* __restrict__ dec_bih,  const float* __restrict__ dec_bhh,
                    const bf16*  __restrict__ fc1W,     const float* __restrict__ fc1_b,
                    const float* __restrict__ fc2_W,    const float* __restrict__ fc2_b,
                    float* __restrict__ out)
{
    __shared__ bf16 hbuf[2][BT][HP];     // ping-pong hidden state (bf16)
    __shared__ bf16 scratch[BT][HP];     // encoder x_t (cols<128) / decoder fc1 activations

    const int tid  = threadIdx.x;
    const int lane = tid & 31;
    const int wave = tid >> 5;
    const int mg   = wave >> 2;          // m-group 0..1: owns M-tiles {2mg, 2mg+1} (32 batch rows)
    const int uo   = wave & 3;           // 4-way N-tile split: u = uo + 4j
    const int ml   = lane & 15;          // M (A) / N (B,C) within tile
    const int lh   = lane >> 4;          // K-half select for A/B; +8 row select for C
    const int rowbase = blockIdx.x * BT;

    // h0 = 0
    for (int i = tid; i < BT * HP; i += NTHREADS) hbuf[0][0][i] = (bf16)0.0f;
    int p = 0;

    // ================= encoder: 9 steps =================
    for (int t = 0; t < 9; ++t) {
        // x_t = tanh(embed) -> scratch (bf16)
        const float* embW = (t < 8) ? emb_v_W : emb_mt_W;
        const float* embB = (t < 8) ? emb_v_b : emb_mt_b;
        const int    xoff = (t < 8) ? (6 + t * 6) : 0;
        for (int idx = tid; idx < BT * EDIM; idx += NTHREADS) {
            int r = idx >> 7, e = idx & 127;
            const float* xr = x + (size_t)(rowbase + r) * 54 + xoff;
            const float* w  = embW + e * 6;
            float acc = embB[e];
            #pragma unroll
            for (int k = 0; k < 6; ++k) acc += xr[k] * w[k];
            scratch[r][e] = (bf16)tanhf_(acc);
        }
        __syncthreads();

        // Preload A fragments for both owned M-tiles: x_t (K=128 -> 4) and h (K=256 -> 8)
        v16bf ax[2][4], ah[2][8];
        #pragma unroll
        for (int mm = 0; mm < 2; ++mm) {
            const int mr = (mg * 2 + mm) * 16 + ml;
            #pragma unroll
            for (int kk = 0; kk < 4; ++kk) ax[mm][kk] = ldfrag(&scratch[mr][kk * 32 + lh * 16]);
            #pragma unroll
            for (int kk = 0; kk < 8; ++kk) ah[mm][kk] = ldfrag(&hbuf[p][mr][kk * 32 + lh * 16]);
        }

        #pragma unroll 1
        for (int j = 0; j < 4; ++j) {
            const int u    = uo + 4 * j;
            const int ncol = u * 16 + ml;
            v8f cir[2] = {{}, {}}, ciz[2] = {{}, {}}, cin_[2] = {{}, {}};
            v8f chr2[2] = {{}, {}}, chz[2] = {{}, {}}, chn[2] = {{}, {}};
            #pragma unroll
            for (int kk = 0; kk < 4; ++kk) {
                const int off = kk * 32 + lh * 16;
                v16bf br = ldfrag(encWih + (size_t)(ncol      ) * EDIM + off);
                v16bf bz = ldfrag(encWih + (size_t)(ncol + 256) * EDIM + off);
                v16bf bn = ldfrag(encWih + (size_t)(ncol + 512) * EDIM + off);
                #pragma unroll
                for (int mm = 0; mm < 2; ++mm) {
                    cir[mm]  = wmma_bf(ax[mm][kk], br, cir[mm]);
                    ciz[mm]  = wmma_bf(ax[mm][kk], bz, ciz[mm]);
                    cin_[mm] = wmma_bf(ax[mm][kk], bn, cin_[mm]);
                }
            }
            #pragma unroll
            for (int kk = 0; kk < 8; ++kk) {
                const int off = kk * 32 + lh * 16;
                v16bf br = ldfrag(encWhh + (size_t)(ncol      ) * HDIM + off);
                v16bf bz = ldfrag(encWhh + (size_t)(ncol + 256) * HDIM + off);
                v16bf bn = ldfrag(encWhh + (size_t)(ncol + 512) * HDIM + off);
                #pragma unroll
                for (int mm = 0; mm < 2; ++mm) {
                    chr2[mm] = wmma_bf(ah[mm][kk], br, chr2[mm]);
                    chz[mm]  = wmma_bf(ah[mm][kk], bz, chz[mm]);
                    chn[mm]  = wmma_bf(ah[mm][kk], bn, chn[mm]);
                }
            }
            const float bir = enc_bih[ncol], biz = enc_bih[ncol + 256], bin_ = enc_bih[ncol + 512];
            const float bhr = enc_bhh[ncol], bhz = enc_bhh[ncol + 256], bhn  = enc_bhh[ncol + 512];
            #pragma unroll
            for (int mm = 0; mm < 2; ++mm) {
                #pragma unroll
                for (int i = 0; i < 8; ++i) {
                    const int r = (mg * 2 + mm) * 16 + i + lh * 8;
                    float rg = sigm_(cir[mm][i] + bir + chr2[mm][i] + bhr);
                    float zg = sigm_(ciz[mm][i] + biz + chz[mm][i] + bhz);
                    float ng = tanhf_(cin_[mm][i] + bin_ + rg * (chn[mm][i] + bhn));
                    float ho = (float)hbuf[p][r][ncol];
                    hbuf[p ^ 1][r][ncol] = (bf16)((1.0f - zg) * ng + zg * ho);
                }
            }
        }
        __syncthreads();
        p ^= 1;
    }

    // ================= decoder: 32 steps (dec_in == h for every step) =================
    for (int s = 0; s < SDEC; ++s) {
        v16bf ah[2][8];
        #pragma unroll
        for (int mm = 0; mm < 2; ++mm) {
            const int mr = (mg * 2 + mm) * 16 + ml;
            #pragma unroll
            for (int kk = 0; kk < 8; ++kk) ah[mm][kk] = ldfrag(&hbuf[p][mr][kk * 32 + lh * 16]);
        }

        #pragma unroll 1
        for (int j = 0; j < 4; ++j) {
            const int u    = uo + 4 * j;
            const int ncol = u * 16 + ml;
            // Warm L2/L1 for the next N-tile's weight rows while this tile computes.
            if (j < 3) {
                const int nc2 = (uo + 4 * (j + 1)) * 16 + ml;
                __builtin_prefetch(decWih + (size_t)(nc2      ) * HDIM, 0, 0);
                __builtin_prefetch(decWih + (size_t)(nc2 + 256) * HDIM, 0, 0);
                __builtin_prefetch(decWih + (size_t)(nc2 + 512) * HDIM, 0, 0);
                __builtin_prefetch(decWhh + (size_t)(nc2      ) * HDIM, 0, 0);
                __builtin_prefetch(decWhh + (size_t)(nc2 + 256) * HDIM, 0, 0);
                __builtin_prefetch(decWhh + (size_t)(nc2 + 512) * HDIM, 0, 0);
            }
            v8f cir[2] = {{}, {}}, ciz[2] = {{}, {}}, cin_[2] = {{}, {}};
            v8f chr2[2] = {{}, {}}, chz[2] = {{}, {}}, chn[2] = {{}, {}};
            #pragma unroll
            for (int kk = 0; kk < 8; ++kk) {
                const int off = kk * 32 + lh * 16;
                v16bf bir_ = ldfrag(decWih + (size_t)(ncol      ) * HDIM + off);
                v16bf biz_ = ldfrag(decWih + (size_t)(ncol + 256) * HDIM + off);
                v16bf bin2 = ldfrag(decWih + (size_t)(ncol + 512) * HDIM + off);
                v16bf bhr_ = ldfrag(decWhh + (size_t)(ncol      ) * HDIM + off);
                v16bf bhz_ = ldfrag(decWhh + (size_t)(ncol + 256) * HDIM + off);
                v16bf bhn_ = ldfrag(decWhh + (size_t)(ncol + 512) * HDIM + off);
                #pragma unroll
                for (int mm = 0; mm < 2; ++mm) {
                    cir[mm]  = wmma_bf(ah[mm][kk], bir_, cir[mm]);
                    ciz[mm]  = wmma_bf(ah[mm][kk], biz_, ciz[mm]);
                    cin_[mm] = wmma_bf(ah[mm][kk], bin2, cin_[mm]);
                    chr2[mm] = wmma_bf(ah[mm][kk], bhr_, chr2[mm]);
                    chz[mm]  = wmma_bf(ah[mm][kk], bhz_, chz[mm]);
                    chn[mm]  = wmma_bf(ah[mm][kk], bhn_, chn[mm]);
                }
            }
            const float bir = dec_bih[ncol], biz = dec_bih[ncol + 256], bin_ = dec_bih[ncol + 512];
            const float bhr = dec_bhh[ncol], bhz = dec_bhh[ncol + 256], bhn  = dec_bhh[ncol + 512];
            #pragma unroll
            for (int mm = 0; mm < 2; ++mm) {
                #pragma unroll
                for (int i = 0; i < 8; ++i) {
                    const int r = (mg * 2 + mm) * 16 + i + lh * 8;
                    float rg = sigm_(cir[mm][i] + bir + chr2[mm][i] + bhr);
                    float zg = sigm_(ciz[mm][i] + biz + chz[mm][i] + bhz);
                    float ng = tanhf_(cin_[mm][i] + bin_ + rg * (chn[mm][i] + bhn));
                    float ho = (float)hbuf[p][r][ncol];
                    hbuf[p ^ 1][r][ncol] = (bf16)((1.0f - zg) * ng + zg * ho);
                }
            }
        }
        __syncthreads();

        // fc1: relu(h_new @ fc1W.T + b1) -> scratch (bf16); one B fetch feeds both M-tiles
        v16bf af[2][8];
        #pragma unroll
        for (int mm = 0; mm < 2; ++mm) {
            const int mr = (mg * 2 + mm) * 16 + ml;
            #pragma unroll
            for (int kk = 0; kk < 8; ++kk) af[mm][kk] = ldfrag(&hbuf[p ^ 1][mr][kk * 32 + lh * 16]);
        }
        #pragma unroll 1
        for (int j = 0; j < 4; ++j) {
            const int nt   = uo + 4 * j;
            const int ncol = nt * 16 + ml;
            v8f c[2] = {{}, {}};
            #pragma unroll
            for (int kk = 0; kk < 8; ++kk) {
                const int off = kk * 32 + lh * 16;
                v16bf b = ldfrag(fc1W + (size_t)ncol * HDIM + off);
                #pragma unroll
                for (int mm = 0; mm < 2; ++mm) c[mm] = wmma_bf(af[mm][kk], b, c[mm]);
            }
            const float b1 = fc1_b[ncol];
            #pragma unroll
            for (int mm = 0; mm < 2; ++mm) {
                #pragma unroll
                for (int i = 0; i < 8; ++i) {
                    const int r = (mg * 2 + mm) * 16 + i + lh * 8;
                    float v = c[mm][i] + b1;
                    scratch[r][ncol] = (bf16)(v > 0.0f ? v : 0.0f);
                }
            }
        }
        __syncthreads();

        // fc2 (OUT=2) + tanh -> out[B, S, 2]
        if (tid < BT * 2) {
            const int r = tid >> 1, o = tid & 1;
            const float* w2 = fc2_W + o * HDIM;
            float acc = fc2_b[o];
            #pragma unroll 4
            for (int k = 0; k < HDIM; ++k) acc += (float)scratch[r][k] * w2[k];
            out[(size_t)(rowbase + r) * (SDEC * 2) + s * 2 + o] = tanhf_(acc);
        }
        __syncthreads();
        p ^= 1;
    }
}

extern "C" void kernel_launch(void* const* d_in, const int* in_sizes, int n_in,
                              void* d_out, int out_size, void* d_ws, size_t ws_size,
                              hipStream_t stream)
{
    const float* x        = (const float*)d_in[0];
    const float* emb_mt_W = (const float*)d_in[1];
    const float* emb_mt_b = (const float*)d_in[2];
    const float* emb_v_W  = (const float*)d_in[3];
    const float* emb_v_b  = (const float*)d_in[4];
    const float* enc_Wih  = (const float*)d_in[5];
    const float* enc_Whh  = (const float*)d_in[6];
    const float* enc_bih  = (const float*)d_in[7];
    const float* enc_bhh  = (const float*)d_in[8];
    const float* dec_Wih  = (const float*)d_in[9];
    const float* dec_Whh  = (const float*)d_in[10];
    const float* dec_bih  = (const float*)d_in[11];
    const float* dec_bhh  = (const float*)d_in[12];
    const float* fc1_W    = (const float*)d_in[13];
    const float* fc1_b    = (const float*)d_in[14];
    const float* fc2_W    = (const float*)d_in[15];
    const float* fc2_b    = (const float*)d_in[16];
    float* out = (float*)d_out;

    // bf16 weight copies in workspace (rebuilt every call: deterministic, graph-safe)
    char* ws = (char*)d_ws;
    bf16* encWihB = (bf16*)(ws + 0);          // 768*128
    bf16* encWhhB = (bf16*)(ws + 196608);     // 768*256
    bf16* decWihB = (bf16*)(ws + 589824);     // 768*256
    bf16* decWhhB = (bf16*)(ws + 983040);     // 768*256
    bf16* fc1WB   = (bf16*)(ws + 1376256);    // 256*256

    cvt_bf16_kernel<<<(98304  + 255) / 256, 256, 0, stream>>>(enc_Wih, encWihB, 98304);
    cvt_bf16_kernel<<<(196608 + 255) / 256, 256, 0, stream>>>(enc_Whh, encWhhB, 196608);
    cvt_bf16_kernel<<<(196608 + 255) / 256, 256, 0, stream>>>(dec_Wih, decWihB, 196608);
    cvt_bf16_kernel<<<(196608 + 255) / 256, 256, 0, stream>>>(dec_Whh, decWhhB, 196608);
    cvt_bf16_kernel<<<(65536  + 255) / 256, 256, 0, stream>>>(fc1_W,   fc1WB,   65536);

    gru_net_kernel<<<B_TOT / BT, NTHREADS, 0, stream>>>(
        x, emb_mt_W, emb_mt_b, emb_v_W, emb_v_b,
        encWihB, encWhhB, enc_bih, enc_bhh,
        decWihB, decWhhB, dec_bih, dec_bhh,
        fc1WB, fc1_b, fc2_W, fc2_b, out);
}
